// HopfieldMemoryBlock_9689446219814
// MI455X (gfx1250) — matrix operand order repaired and checked by reading, not compiled, not run on hardware
//
#include <hip/hip_runtime.h>
#include <hip/hip_bf16.h>
#include <math.h>

typedef __bf16 bf16;
typedef __attribute__((ext_vector_type(16))) __bf16 v16bf;
typedef __attribute__((ext_vector_type(8)))  float  v8f;

union FragU { v16bf v; uint4 q[2]; };
static_assert(sizeof(v16bf) == 32, "v16bf must be 32 bytes");

#define C_DIM 256
#define M_DIM 64
#define T_DIM 64
#define NPIX  4096
#define LDZF  260   // fp32 z master row stride (floats)
#define LDZB  264   // bf16 z_norm / x tile row stride (bf16)
#define LDSB  68    // fp32 logits row stride (floats)
#define LDAT  72    // bf16 attn row stride (bf16)
#define BETA  0.5f

// ---- WMMA fragment loaders (ISA 7.12.2 16-bit layouts) ------------------
// A 16x32: lane r<16 -> M=r, K in {0..7}U{16..23}; lanes 16..31 -> K {8..15}U{24..31}
__device__ __forceinline__ v16bf frag_a(const bf16* p, int ld, int lane) {
  int r = lane & 15, h = lane >> 4;
  const bf16* q = p + r * ld + h * 8;
  FragU u;
  u.q[0] = *(const uint4*)(q);        // K = h*8 .. h*8+7
  u.q[1] = *(const uint4*)(q + 16);   // K = 16+h*8 .. 16+h*8+7
  return u.v;
}
// B 32x16 from row-major N x K storage: lane r<16 -> N=r, K {0..7}U{16..23}; lanes 16..31 -> K {8..15}U{24..31}
__device__ __forceinline__ v16bf frag_b(const bf16* p, int ld, int lane) {
  int r = lane & 15, h = lane >> 4;
  const bf16* q = p + r * ld + h * 8;
  FragU u;
  u.q[0] = *(const uint4*)(q);        // K = h*8 .. h*8+7
  u.q[1] = *(const uint4*)(q + 16);   // K = 16+h*8 .. 16+h*8+7
  return u.v;
}
#define WMMA_BF16(A, B, Cacc) \
  __builtin_amdgcn_wmma_f32_16x16x32_bf16(false, (A), false, (B), (short)0, (Cacc), false, false)

// ---- prep: normalize memory bank rows; emit bf16 mem_norm and mem_norm^T ----
__global__ __launch_bounds__(256)
void prep_mem_kernel(const float* __restrict__ memb,
                     bf16* __restrict__ memnA,   // 64 x 256 (row m, contiguous c)
                     bf16* __restrict__ memnT) { // 256 x 64 (row c, contiguous m)
  int tid  = threadIdx.x;
  int row  = tid >> 2, part = tid & 3;
  const float* src = memb + row * C_DIM + part * 64;
  float ss = 0.f;
  #pragma unroll 8
  for (int c = 0; c < 64; ++c) { float v = src[c]; ss += v * v; }
  ss += __shfl_xor(ss, 1);
  ss += __shfl_xor(ss, 2);
  float invn = 1.0f / fmaxf(sqrtf(ss), 1e-12f);
  for (int c = 0; c < 64; ++c) {
    float v = src[c] * invn;
    int cc = part * 64 + c;
    memnA[row * C_DIM + cc] = (bf16)v;
    memnT[cc * M_DIM + row] = (bf16)v;
  }
}

// ---- prep: fp32 -> bf16 weight conversion -------------------------------
__global__ __launch_bounds__(256)
void convert_w_kernel(const float* __restrict__ Wq, const float* __restrict__ Wout,
                      bf16* __restrict__ WqB, bf16* __restrict__ WoutB, int n) {
  for (int i = blockIdx.x * blockDim.x + threadIdx.x; i < 2 * n;
       i += gridDim.x * blockDim.x) {
    if (i < n) WqB[i]       = (bf16)Wq[i];
    else       WoutB[i - n] = (bf16)Wout[i - n];
  }
}

// ---- fused Hopfield block: 64 tokens per workgroup, everything in LDS ----
__global__ __launch_bounds__(256)
void hopfield_fused_kernel(const float* __restrict__ x,
                           const float* __restrict__ bq,
                           const float* __restrict__ bout,
                           const bf16* __restrict__ WqB,
                           const bf16* __restrict__ WoutB,
                           const bf16* __restrict__ memnA,
                           const bf16* __restrict__ memnT,
                           float* __restrict__ out) {
  extern __shared__ char smem[];
  float* Zf = reinterpret_cast<float*>(smem);                                        // 66560 B
  bf16*  Zb = reinterpret_cast<bf16*>(smem + T_DIM * LDZF * 4);                      // 33792 B
  float* Sb = reinterpret_cast<float*>(smem + T_DIM * LDZF * 4 + T_DIM * LDZB * 2);  // 17408 B
  bf16*  At = reinterpret_cast<bf16*>(smem + T_DIM * LDZF * 4 + T_DIM * LDZB * 2 +
                                      T_DIM * LDSB * 4);                             //  9216 B

  const int tid  = threadIdx.x;
  const int lane = tid & 31;
  const int wave = tid >> 5;
  const int n16  = lane & 15;
  const int h16  = lane >> 4;

  const int bimg = blockIdx.x >> 6;
  const int t0   = (blockIdx.x & 63) * T_DIM;
  const float* xb = x   + (size_t)bimg * C_DIM * NPIX + t0;
  float*       ob = out + (size_t)bimg * C_DIM * NPIX + t0;

  // load x tile (C x T, strided rows in global) -> Zb as T x C bf16 (coalesced reads)
  for (int idx = tid; idx < C_DIM * T_DIM; idx += 256) {
    int c = idx >> 6, t = idx & 63;
    Zb[t * LDZB + c] = (bf16)xb[(size_t)c * NPIX + t];
  }
  __syncthreads();

  v8f acc[2][4];
  const v8f vzero = {0.f, 0.f, 0.f, 0.f, 0.f, 0.f, 0.f, 0.f};

  // D[c-tile(32) x t-tile(64)] = A(global, 256xK=256 row-major) * Zb; per wave
  auto gemm_cxk256 = [&](const bf16* __restrict__ A) {
    #pragma unroll
    for (int oi = 0; oi < 2; ++oi)
      #pragma unroll
      for (int ti = 0; ti < 4; ++ti) acc[oi][ti] = vzero;
    const bf16* A0 = A + (wave * 32)      * C_DIM;
    const bf16* A1 = A + (wave * 32 + 16) * C_DIM;
    for (int k = 0; k < C_DIM; k += 32) {
      v16bf a0 = frag_a(A0 + k, C_DIM, lane);
      v16bf a1 = frag_a(A1 + k, C_DIM, lane);
      #pragma unroll
      for (int ti = 0; ti < 4; ++ti) {
        v16bf bb = frag_b(Zb + (ti * 16) * LDZB + k, LDZB, lane);
        acc[0][ti] = WMMA_BF16(a0, bb, acc[0][ti]);
        acc[1][ti] = WMMA_BF16(a1, bb, acc[1][ti]);
      }
    }
  };

  // ---- phase A: q = Wq x + bq -> Zf ----
  gemm_cxk256(WqB);
  #pragma unroll
  for (int oi = 0; oi < 2; ++oi) {
    int cbase = wave * 32 + oi * 16 + 8 * h16;
    float bqv[8];
    #pragma unroll
    for (int v = 0; v < 8; ++v) bqv[v] = bq[cbase + v];
    #pragma unroll
    for (int ti = 0; ti < 4; ++ti) {
      int t = ti * 16 + n16;
      #pragma unroll
      for (int v = 0; v < 8; ++v)
        Zf[t * LDZF + cbase + v] = acc[oi][ti][v] + bqv[v];
    }
  }
  __syncthreads();

  // per-token L2 norm (4 lanes per token), writes bf16 z_norm into Zb
  auto l2normalize = [&](bool update_master) {
    int t = tid >> 2, part = tid & 3;
    float* row = Zf + t * LDZF + part * 64;
    bf16*  brw = Zb + t * LDZB + part * 64;
    float ss = 0.f;
    #pragma unroll 8
    for (int c = 0; c < 64; ++c) { float v = row[c]; ss += v * v; }
    ss += __shfl_xor(ss, 1);
    ss += __shfl_xor(ss, 2);
    float invn = 1.0f / fmaxf(sqrtf(ss), 1e-12f);
    #pragma unroll 8
    for (int c = 0; c < 64; ++c) {
      float v = row[c] * invn;
      brw[c] = (bf16)v;
      if (update_master) row[c] = v;
    }
  };

  l2normalize(true);  // z = q_norm; Zb now holds z_norm for iteration 0
  __syncthreads();

  for (int it = 0; it < 3; ++it) {
    if (it) { l2normalize(false); __syncthreads(); }

    // logits S[m, t] = mem_norm[m,:] . z_norm[:, t]  (64 x 64, 16 tiles / 8 waves)
    {
      int mb  = wave & 3;
      int tb0 = (wave >> 2) * 2;
      v8f s0 = vzero, s1 = vzero;
      const bf16* Am = memnA + (mb * 16) * C_DIM;
      for (int k = 0; k < C_DIM; k += 32) {
        v16bf a  = frag_a(Am + k, C_DIM, lane);
        v16bf b0 = frag_b(Zb + (tb0 * 16) * LDZB + k, LDZB, lane);
        v16bf b1 = frag_b(Zb + (tb0 * 16 + 16) * LDZB + k, LDZB, lane);
        s0 = WMMA_BF16(a, b0, s0);
        s1 = WMMA_BF16(a, b1, s1);
      }
      int mrow = mb * 16 + 8 * h16;
      #pragma unroll
      for (int v = 0; v < 8; ++v) {
        Sb[(tb0 * 16 + n16) * LDSB + mrow + v]      = s0[v];
        Sb[(tb0 * 16 + 16 + n16) * LDSB + mrow + v] = s1[v];
      }
    }
    __syncthreads();

    // softmax over m=64 per token -> bf16 attn (T x M)
    {
      int t = tid >> 2, part = tid & 3;
      const float* srow = Sb + t * LDSB + part * 16;
      float mx = -3.402823466e38f;
      #pragma unroll
      for (int m = 0; m < 16; ++m) mx = fmaxf(mx, srow[m]);
      mx = fmaxf(mx, __shfl_xor(mx, 1));
      mx = fmaxf(mx, __shfl_xor(mx, 2));
      float e[16]; float sum = 0.f;
      #pragma unroll
      for (int m = 0; m < 16; ++m) { e[m] = __expf(srow[m] - mx); sum += e[m]; }
      sum += __shfl_xor(sum, 1);
      sum += __shfl_xor(sum, 2);
      float inv = 1.0f / sum;
      bf16* arow = At + t * LDAT + part * 16;
      #pragma unroll
      for (int m = 0; m < 16; ++m) arow[m] = (bf16)(e[m] * inv);
    }
    __syncthreads();

    // m_agg[c, t] = sum_m mem_norm^T[c, m] * attn[m, t]; z = (1-beta) z + beta m_agg
    {
      v8f g[2][4];
      #pragma unroll
      for (int oi = 0; oi < 2; ++oi)
        #pragma unroll
        for (int ti = 0; ti < 4; ++ti) g[oi][ti] = vzero;
      const bf16* A0 = memnT + (wave * 32)      * M_DIM;
      const bf16* A1 = memnT + (wave * 32 + 16) * M_DIM;
      #pragma unroll
      for (int k = 0; k < M_DIM; k += 32) {
        v16bf a0 = frag_a(A0 + k, M_DIM, lane);
        v16bf a1 = frag_a(A1 + k, M_DIM, lane);
        #pragma unroll
        for (int ti = 0; ti < 4; ++ti) {
          v16bf bb = frag_b(At + (ti * 16) * LDAT + k, LDAT, lane);
          g[0][ti] = WMMA_BF16(a0, bb, g[0][ti]);
          g[1][ti] = WMMA_BF16(a1, bb, g[1][ti]);
        }
      }
      #pragma unroll
      for (int oi = 0; oi < 2; ++oi) {
        int cbase = wave * 32 + oi * 16 + 8 * h16;
        #pragma unroll
        for (int ti = 0; ti < 4; ++ti) {
          int t = ti * 16 + n16;
          #pragma unroll
          for (int v = 0; v < 8; ++v) {
            float* zp = Zf + t * LDZF + cbase + v;
            *zp = (1.0f - BETA) * (*zp) + BETA * g[oi][ti][v];
          }
        }
      }
    }
    __syncthreads();
  }

  // ---- epilogue: out = x + Wout z + bout ----
  for (int idx = tid; idx < T_DIM * C_DIM; idx += 256) {
    int t = idx >> 8, c = idx & 255;
    Zb[t * LDZB + c] = (bf16)Zf[t * LDZF + c];
  }
  __syncthreads();

  gemm_cxk256(WoutB);
  #pragma unroll
  for (int oi = 0; oi < 2; ++oi) {
    int cbase = wave * 32 + oi * 16 + 8 * h16;
    float bov[8];
    #pragma unroll
    for (int v = 0; v < 8; ++v) bov[v] = bout[cbase + v];
    #pragma unroll
    for (int ti = 0; ti < 4; ++ti) {
      int t = ti * 16 + n16;
      #pragma unroll
      for (int v = 0; v < 8; ++v) {
        size_t off = (size_t)(cbase + v) * NPIX + t;
        ob[off] = xb[off] + acc[oi][ti][v] + bov[v];
      }
    }
  }
}

extern "C" void kernel_launch(void* const* d_in, const int* in_sizes, int n_in,
                              void* d_out, int out_size, void* d_ws, size_t ws_size,
                              hipStream_t stream) {
  const float* x    = (const float*)d_in[0];
  const float* memb = (const float*)d_in[1];
  const float* Wq   = (const float*)d_in[2];
  const float* bq   = (const float*)d_in[3];
  const float* Wout = (const float*)d_in[4];
  const float* bout = (const float*)d_in[5];
  float* out = (float*)d_out;

  char* ws = (char*)d_ws;
  bf16* WqB   = (bf16*)(ws);            // 256*256*2 = 131072 B
  bf16* WoutB = (bf16*)(ws + 131072);   // 131072 B
  bf16* memnA = (bf16*)(ws + 262144);   // 64*256*2 = 32768 B
  bf16* memnT = (bf16*)(ws + 294912);   // 256*64*2 = 32768 B

  int B = in_sizes[0] / (C_DIM * NPIX);

  convert_w_kernel<<<256, 256, 0, stream>>>(Wq, Wout, WqB, WoutB, C_DIM * C_DIM);
  prep_mem_kernel<<<1, 256, 0, stream>>>(memb, memnA, memnT);

  constexpr int SMEM_BYTES =
      T_DIM * LDZF * 4 + T_DIM * LDZB * 2 + T_DIM * LDSB * 4 + T_DIM * LDAT * 2; // 126976
  (void)hipFuncSetAttribute((const void*)hopfield_fused_kernel,
                            hipFuncAttributeMaxDynamicSharedMemorySize, SMEM_BYTES);
  hopfield_fused_kernel<<<dim3(B * 64), dim3(256), SMEM_BYTES, stream>>>(
      x, bq, bout, WqB, WoutB, memnA, memnT, out);
}
